// AttentionBlock_69363721831003
// MI455X (gfx1250) — compile-verified, matrix operand order checked by reading, not run on hardware
//
#include <hip/hip_runtime.h>
#include <hip/hip_bf16.h>

typedef __attribute__((ext_vector_type(16))) __bf16 bf16x16;
typedef __attribute__((ext_vector_type(8)))  float  f32x8;
typedef __attribute__((ext_vector_type(4)))  unsigned int u32x4;
typedef __attribute__((ext_vector_type(8)))  int  i32x8;
typedef __attribute__((ext_vector_type(4)))  int  i32x4;

#define BB 4
#define SS 2048
#define EE 1024
#define HH 16
#define DD 64
#define BS (BB*SS)      // 8192 rows
#define NQKV (3*HH*DD)  // 3072
#define HD (HH*DD)      // 1024

// ---- TDM availability ----------------------------------------------------
#if defined(__has_builtin)
#if __has_builtin(__builtin_amdgcn_tensor_load_to_lds) && __has_builtin(__builtin_amdgcn_s_wait_tensorcnt)
#define USE_TDM 1
#endif
#endif
#ifndef USE_TDM
#define USE_TDM 0
#endif

// ---- helpers -------------------------------------------------------------

static __device__ inline f32x8 zero8() {
  f32x8 z;
#pragma unroll
  for (int i = 0; i < 8; ++i) z[i] = 0.0f;
  return z;
}

// Build a 16x32 bf16 A-fragment (or a 32x16 B-fragment read as rows of B^T)
// from two 16-byte chunks: elements 0..7 from p0, 8..15 from p1.
static __device__ inline bf16x16 load_frag(const __bf16* p0, const __bf16* p1) {
  bf16x16 r;
  *(uint4*)&r       = *(const uint4*)p0;
  *((uint4*)&r + 1) = *(const uint4*)p1;
  return r;
}

#if USE_TDM
// Issue a TDM 2D tile load: tile_w x tile_h bf16 elements from a row-major
// tensor (row length tensor_w, row stride `stride`, tensor_h rows) into LDS.
// D# layout per CDNA5 ISA 8.3/8.4: group0 = {flags, lds_addr, global_addr, type},
// group1 = {wg_mask/data_size, tensor dims, tile dims, strides}.
static __device__ inline void tdm_load_2d(unsigned lds_off, const __bf16* gptr,
                                          unsigned tensor_w, unsigned tensor_h,
                                          unsigned stride_elems,
                                          unsigned tile_w, unsigned tile_h) {
  unsigned long long ga = (unsigned long long)(size_t)gptr;
  u32x4 g0;
  g0[0] = 1u;                                        // count=1, user-mode D#
  g0[1] = lds_off;                                   // lds_addr (bytes)
  g0[2] = (unsigned)(ga & 0xffffffffu);              // global_addr[31:0]
  g0[3] = (unsigned)((ga >> 32) & 0x01ffffffu)       // global_addr[56:32]
        | (2u << 30);                                // type = 2 ("image")
  i32x8 g1;
  g1[0] = (1 << 16);                                 // data_size=1 (2B), wg_mask=0
  g1[1] = (int)((tensor_w & 0xffffu) << 16);         // abar_addr=0 | tensor_dim0 lo
  g1[2] = (int)(((tensor_w >> 16) & 0xffffu) | ((tensor_h & 0xffffu) << 16));
  g1[3] = (int)(((tensor_h >> 16) & 0xffffu) | ((tile_w & 0xffffu) << 16));
  g1[4] = (int)(tile_h & 0xffffu);                   // tile_dim1 | tile_dim2=0
  g1[5] = (int)stride_elems;                         // tensor_dim0_stride[31:0]
  g1[6] = 0;                                         // stride hi | dim1_stride lo
  g1[7] = 0;
  i32x4 gz; gz[0] = 0; gz[1] = 0; gz[2] = 0; gz[3] = 0;
#if __clang_major__ >= 23
  i32x8 gz8;
#pragma unroll
  for (int i = 0; i < 8; ++i) gz8[i] = 0;
  __builtin_amdgcn_tensor_load_to_lds(g0, g1, gz, gz, gz8, 0);
#else
  __builtin_amdgcn_tensor_load_to_lds(g0, g1, gz, gz, 0);
#endif
}
#endif  // USE_TDM

// ---- prep kernels --------------------------------------------------------

__global__ void k_cvt_bf16(const float* __restrict__ in, __bf16* __restrict__ out, int n) {
  int i = blockIdx.x * blockDim.x + threadIdx.x;
  if (i < n) out[i] = (__bf16)in[i];
}

// out[n*K + k] = (bf16) in[k*N + n]   (transpose + convert)
__global__ void k_trcvt_bf16(const float* __restrict__ in, __bf16* __restrict__ out,
                             int K, int N) {
  int i = blockIdx.x * blockDim.x + threadIdx.x;
  if (i >= N * K) return;
  int n = i / K, k = i - n * K;
  out[i] = (__bf16)in[(size_t)k * N + n];
}

// ---- generic mainloop body (K-step 64, double-buffered LDS B tile) -------
// Wave tile 16(M) x 128(N); 16 WMMAs per k-step flattened over (chunk, ns)
// with an explicit bcur/bnext software pipeline so each fragment's ds_load
// issues before the previous fragment's WMMA (keeps DScnt > 0 at the waits).

#define GEMM_MAINLOOP(AROW, WTILE, KDIM, TENSOR_H)                                   \
  do {                                                                               \
    const int NK = (KDIM) / 64;                                                      \
    bf16x16 a0 = load_frag((AROW) + hi * 8,      (AROW) + 16 + hi * 8);              \
    bf16x16 a1 = load_frag((AROW) + 32 + hi * 8, (AROW) + 48 + hi * 8);              \
    for (int i = 0; i < NK; ++i) {                                                   \
      const int k0 = i * 64;                                                         \
      TILE_FETCH(i, k0, KDIM, TENSOR_H, WTILE);                                      \
      bf16x16 a0n = a0, a1n = a1;                                                    \
      if (i + 1 < NK) {                                                              \
        a0n = load_frag((AROW) + k0 + 64 + hi * 8, (AROW) + k0 + 80 + hi * 8);       \
        a1n = load_frag((AROW) + k0 + 96 + hi * 8, (AROW) + k0 + 112 + hi * 8);      \
      }                                                                              \
      const __bf16* bb = &Bt[i & 1][0];                                              \
      bf16x16 bcur = load_frag(bb + lo * 64 + hi * 8, bb + lo * 64 + 16 + hi * 8);   \
      _Pragma("unroll")                                                              \
      for (int u = 0; u < 16; ++u) {                                                 \
        const int c = u >> 3, ns = u & 7;                                            \
        bf16x16 bnext = bcur;                                                        \
        if (u + 1 < 16) {                                                            \
          const int c2 = (u + 1) >> 3, ns2 = (u + 1) & 7;                            \
          const __bf16* br = bb + (ns2 * 16 + lo) * 64 + c2 * 32;                    \
          bnext = load_frag(br + hi * 8, br + 16 + hi * 8);                          \
        }                                                                            \
        acc[ns] = __builtin_amdgcn_wmma_f32_16x16x32_bf16(                           \
            false, c ? a1 : a0, false, bcur, (short)0, acc[ns], false, false);       \
        bcur = bnext;                                                                \
      }                                                                              \
      a0 = a0n; a1 = a1n;                                                            \
      TILE_RELEASE();                                                                \
    }                                                                                \
  } while (0)

#if USE_TDM
#define TILE_FETCH(i, k0, KDIM, TENSOR_H, WTILE)                                     \
  do {                                                                               \
    if (wave == 0) {                                                                 \
      if ((i) + 1 < (KDIM) / 64) {                                                   \
        tdm_load_2d(lds_b[((i) + 1) & 1], (WTILE) + (k0) + 64, (KDIM), (TENSOR_H),   \
                    (KDIM), 64, 128);                                                \
        __builtin_amdgcn_s_wait_tensorcnt(1);                                        \
      } else {                                                                       \
        __builtin_amdgcn_s_wait_tensorcnt(0);                                        \
      }                                                                              \
    }                                                                                \
    __syncthreads();                                                                 \
  } while (0)
#define TILE_RELEASE() __syncthreads()
#else
#define TILE_FETCH(i, k0, KDIM, TENSOR_H, WTILE)                                     \
  do {                                                                               \
    __syncthreads();                                                                 \
    _Pragma("unroll")                                                                \
    for (int c_ = 0; c_ < 4; ++c_) {                                                 \
      int chunk = t + c_ * 256, row = chunk >> 3, seg = chunk & 7;                   \
      *(uint4*)&Bt[(i) & 1][row * 64 + seg * 8] =                                    \
          *(const uint4*)((WTILE) + (size_t)row * (KDIM) + (k0) + seg * 8);          \
    }                                                                                \
    __syncthreads();                                                                 \
  } while (0)
#define TILE_RELEASE() ((void)0)
#endif

// ---- QKV projection GEMM: (8192x1024) x (1024x3072) ----------------------

__global__ __launch_bounds__(256)
void k_qkv_gemm(const __bf16* __restrict__ xb, const __bf16* __restrict__ wT,
                const float* __restrict__ bias,
                __bf16* __restrict__ q, __bf16* __restrict__ kk, __bf16* __restrict__ vT) {
  __shared__ __align__(16) __bf16 Bt[2][128 * 64];   // 2 x 16 KB
  const int t = threadIdx.x;
  const int wave = t >> 5, lane = t & 31, lo = lane & 15, hi = lane >> 4;
  const int m0 = blockIdx.x * 128 + wave * 16;
  const int n0 = blockIdx.y * 128;

  f32x8 acc[8];
#pragma unroll
  for (int j = 0; j < 8; ++j) acc[j] = zero8();

  const __bf16* xrow = xb + (size_t)(m0 + lo) * EE;
  const __bf16* wtile = wT + (size_t)n0 * EE;

#if USE_TDM
  const unsigned lds_b[2] = { (unsigned)(size_t)&Bt[0][0], (unsigned)(size_t)&Bt[1][0] };
  if (wave == 0)
    tdm_load_2d(lds_b[0], wtile, EE, NQKV, EE, 64, 128);
#endif

  GEMM_MAINLOOP(xrow, wtile, EE, NQKV);

#pragma unroll
  for (int ns = 0; ns < 8; ++ns) {
    int n = n0 + ns * 16 + lo;
    int tt = n >> 10, rem = n & 1023, h = rem >> 6, d = rem & 63;
    float bv = bias[n];
#pragma unroll
    for (int r = 0; r < 8; ++r) {
      int m = m0 + r + 8 * hi;
      int bidx = m >> 11, s = m & (SS - 1);
      __bf16 val = (__bf16)(acc[ns][r] + bv);
      if (tt == 0)      q [((size_t)(bidx * HH + h) * SS + s) * DD + d] = val;
      else if (tt == 1) kk[((size_t)(bidx * HH + h) * SS + s) * DD + d] = val;
      else              vT[((size_t)(bidx * HH + h) * DD + d) * SS + s] = val;
    }
  }
}

// ---- causal flash attention ---------------------------------------------
// grid: (S/64, B*H), block 128 (4 waves, 16 q-rows each). Barrier-free inner
// loop: per-wave LDS P staging relies on in-order DS within a wave. V
// fragments are hoisted above the softmax so their global latency is covered
// by the VALU-heavy online-softmax block.

__global__ __launch_bounds__(128)
void k_flash_attn(const __bf16* __restrict__ q, const __bf16* __restrict__ kk,
                  const __bf16* __restrict__ vT, __bf16* __restrict__ ans) {
  __shared__ __align__(16) __bf16 Pbuf[4][16 * 32];
  const int t = threadIdx.x;
  const int wave = t >> 5, lane = t & 31, lo = lane & 15, hi = lane >> 4;
  const int bh = blockIdx.y;
  const int q0 = blockIdx.x * 64 + wave * 16;
  const size_t base = (size_t)bh * SS * DD;

  const __bf16* qrow = q + base + (size_t)(q0 + lo) * DD;
  bf16x16 aq0 = load_frag(qrow + hi * 8,      qrow + 16 + hi * 8);
  bf16x16 aq1 = load_frag(qrow + 32 + hi * 8, qrow + 48 + hi * 8);

  float mrow[8], lrow[8];
#pragma unroll
  for (int r = 0; r < 8; ++r) { mrow[r] = -1e30f; lrow[r] = 0.0f; }
  f32x8 acc[4];
#pragma unroll
  for (int j = 0; j < 4; ++j) acc[j] = zero8();

  __bf16* pb = &Pbuf[wave][0];
  const int nkt = (q0 + 47) >> 5;  // key tiles of 32 covering keys <= q0+15

  for (int kt = 0; kt < nkt; ++kt) {
    const int s0 = kt * 32;
    // ---- S = Q K^T over this 16x32 key tile; interleave the two C-chains
    f32x8 sc0 = zero8(), sc1 = zero8();
    {
      const __bf16* krow  = kk + base + (size_t)(s0 + lo) * DD;
      const __bf16* krow2 = krow + 16 * DD;
      bf16x16 b0 = load_frag(krow  + hi * 8,      krow  + 16 + hi * 8);
      bf16x16 b2 = load_frag(krow2 + hi * 8,      krow2 + 16 + hi * 8);
      bf16x16 b1 = load_frag(krow  + 32 + hi * 8, krow  + 48 + hi * 8);
      bf16x16 b3 = load_frag(krow2 + 32 + hi * 8, krow2 + 48 + hi * 8);
      sc0 = __builtin_amdgcn_wmma_f32_16x16x32_bf16(false, aq0, false, b0, (short)0, sc0, false, false);
      sc1 = __builtin_amdgcn_wmma_f32_16x16x32_bf16(false, aq0, false, b2, (short)0, sc1, false, false);
      sc0 = __builtin_amdgcn_wmma_f32_16x16x32_bf16(false, aq1, false, b1, (short)0, sc0, false, false);
      sc1 = __builtin_amdgcn_wmma_f32_16x16x32_bf16(false, aq1, false, b3, (short)0, sc1, false, false);
    }
    // ---- V fragments for this tile: issue now, consume after softmax
    bf16x16 vf[4];
#pragma unroll
    for (int ds_ = 0; ds_ < 4; ++ds_) {
      const __bf16* vrow = vT + ((size_t)bh * DD + ds_ * 16 + lo) * SS + s0;
      vf[ds_] = load_frag(vrow + hi * 8, vrow + 16 + hi * 8);
    }
    // ---- online softmax (rows live in VGPR index, cols across 16 lanes)
#pragma unroll
    for (int r = 0; r < 8; ++r) {
      const int mg = q0 + r + 8 * hi;
      float v0 = sc0[r] * 0.125f; if (s0 + lo      > mg) v0 = -1e30f;
      float v1 = sc1[r] * 0.125f; if (s0 + 16 + lo > mg) v1 = -1e30f;
      float rm = fmaxf(v0, v1);
      rm = fmaxf(rm, __shfl_xor(rm, 1, 32));
      rm = fmaxf(rm, __shfl_xor(rm, 2, 32));
      rm = fmaxf(rm, __shfl_xor(rm, 4, 32));
      rm = fmaxf(rm, __shfl_xor(rm, 8, 32));
      const float mn = fmaxf(mrow[r], rm);
      const float e0 = __expf(v0 - mn), e1 = __expf(v1 - mn);
      float rs = e0 + e1;
      rs += __shfl_xor(rs, 1, 32);
      rs += __shfl_xor(rs, 2, 32);
      rs += __shfl_xor(rs, 4, 32);
      rs += __shfl_xor(rs, 8, 32);
      const float alpha = __expf(mrow[r] - mn);
      lrow[r] = lrow[r] * alpha + rs;
      mrow[r] = mn;
#pragma unroll
      for (int j = 0; j < 4; ++j) acc[j][r] *= alpha;
      pb[(r + 8 * hi) * 32 + lo]      = (__bf16)e0;
      pb[(r + 8 * hi) * 32 + 16 + lo] = (__bf16)e1;
    }
    // reload P as an A-fragment (16x32) and accumulate P * V^T-tile
    bf16x16 pa = load_frag(pb + lo * 32 + hi * 8, pb + lo * 32 + 16 + hi * 8);
#pragma unroll
    for (int ds_ = 0; ds_ < 4; ++ds_) {
      acc[ds_] = __builtin_amdgcn_wmma_f32_16x16x32_bf16(false, pa, false, vf[ds_],
                                                         (short)0, acc[ds_], false, false);
    }
  }

  // ---- epilogue: normalize and write answers as [b][s][h*64+d] bf16
  const int bb = bh >> 4, h = bh & 15;
#pragma unroll
  for (int r = 0; r < 8; ++r) {
    const float inv = 1.0f / lrow[r];
    const int s = q0 + r + 8 * hi;
    const size_t rowo = ((size_t)(bb * SS + s)) * HD + h * DD;
#pragma unroll
    for (int j = 0; j < 4; ++j)
      ans[rowo + j * 16 + lo] = (__bf16)(acc[j][r] * inv);
  }
}

// ---- output projection: (8192x1024) x (1024x1024) + bias, fp32 out -------

__global__ __launch_bounds__(256)
void k_out_gemm(const __bf16* __restrict__ ansb, const __bf16* __restrict__ wT,
                const float* __restrict__ bias, float* __restrict__ out) {
  __shared__ __align__(16) __bf16 Bt[2][128 * 64];
  const int t = threadIdx.x;
  const int wave = t >> 5, lane = t & 31, lo = lane & 15, hi = lane >> 4;
  const int m0 = blockIdx.x * 128 + wave * 16;
  const int n0 = blockIdx.y * 128;

  f32x8 acc[8];
#pragma unroll
  for (int j = 0; j < 8; ++j) acc[j] = zero8();

  const __bf16* arow = ansb + (size_t)(m0 + lo) * HD;
  const __bf16* wtile = wT + (size_t)n0 * HD;

#if USE_TDM
  const unsigned lds_b[2] = { (unsigned)(size_t)&Bt[0][0], (unsigned)(size_t)&Bt[1][0] };
  if (wave == 0)
    tdm_load_2d(lds_b[0], wtile, HD, EE, HD, 64, 128);
#endif

  GEMM_MAINLOOP(arow, wtile, HD, EE);

#pragma unroll
  for (int ns = 0; ns < 8; ++ns) {
    const int e = n0 + ns * 16 + lo;
    const float bv = bias[e];
#pragma unroll
    for (int r = 0; r < 8; ++r) {
      const int m = m0 + r + 8 * hi;
      out[(size_t)m * EE + e] = acc[ns][r] + bv;
    }
  }
}

// ---- launcher ------------------------------------------------------------

extern "C" void kernel_launch(void* const* d_in, const int* in_sizes, int n_in,
                              void* d_out, int out_size, void* d_ws, size_t ws_size,
                              hipStream_t stream) {
  const float* x     = (const float*)d_in[0];
  const float* W_qkv = (const float*)d_in[1];
  const float* b_qkv = (const float*)d_in[2];
  const float* W_out = (const float*)d_in[3];
  const float* b_out = (const float*)d_in[4];
  float* out = (float*)d_out;

  char* ws = (char*)d_ws;
  const size_t MB = 1u << 20;
  __bf16* xb    = (__bf16*)(ws);            // 16 MB  x as bf16 [8192][1024]
  __bf16* wqkvT = (__bf16*)(ws + 16 * MB);  //  6 MB  W_qkv^T  [3072][1024]
  __bf16* woutT = (__bf16*)(ws + 22 * MB);  //  2 MB  W_out^T  [1024][1024]
  __bf16* q     = (__bf16*)(ws + 24 * MB);  // 16 MB  [b][h][s][d]
  __bf16* kk    = (__bf16*)(ws + 40 * MB);  // 16 MB  [b][h][s][d]
  __bf16* vT    = (__bf16*)(ws + 56 * MB);  // 16 MB  [b][h][d][s]
  __bf16* ans   = (__bf16*)(ws + 72 * MB);  // 16 MB  [b*s][h*d]   (ends 88 MB)

  {
    int n = BS * EE;
    k_cvt_bf16<<<dim3((n + 255) / 256), 256, 0, stream>>>(x, xb, n);
    int nq = NQKV * EE;
    k_trcvt_bf16<<<dim3((nq + 255) / 256), 256, 0, stream>>>(W_qkv, wqkvT, EE, NQKV);
    int no = EE * HD;
    k_trcvt_bf16<<<dim3((no + 255) / 256), 256, 0, stream>>>(W_out, woutT, HD, EE);
  }
  k_qkv_gemm<<<dim3(BS / 128, NQKV / 128), 256, 0, stream>>>(xb, wqkvT, b_qkv, q, kk, vT);
  k_flash_attn<<<dim3(SS / 64, BB * HH), 128, 0, stream>>>(q, kk, vT, ans);
  k_out_gemm<<<dim3(BS / 128, EE / 128), 256, 0, stream>>>(ans, woutT, b_out, out);
}
